// TransformerDecoder_53807350284900
// MI455X (gfx1250) — compile-verified, matrix operand order checked by reading, not run on hardware
//
#include <hip/hip_runtime.h>

// ---------------------------------------------------------------------------
// Types for CDNA5 WMMA (wave32, 16x16x32 bf16 -> f32)
// ---------------------------------------------------------------------------
typedef __bf16 bf16_t;
typedef __attribute__((ext_vector_type(16))) __bf16 v16bf;
typedef __attribute__((ext_vector_type(8)))  float  v8f;

#define DMODEL 512
#define NHEAD  8
#define HDIM   64
#define DFFN   2048
#define VOCAB  128
#define NB     64      // batch
#define SEQ    512     // encoder sequence (8*64 spatial)
#define TDEC   48      // seq_size
#define LENC   3
#define LDEC   3
#define LDSP   40      // padded LDS K-stride (16B-aligned rows, conflict-free)

// ---------------------------------------------------------------------------
// Sinusoidal positional encoding value (interleaved sin/cos), computed inline
// ---------------------------------------------------------------------------
__device__ __forceinline__ float pe_val(int pos, int d, int D) {
  int i = d >> 1;
  float div = __expf((float)(2 * i) * (-9.210340371976184f / (float)D));
  float ang = (float)pos * div;
  return (d & 1) ? __cosf(ang) : __sinf(ang);
}

// ---------------------------------------------------------------------------
// Generic batched bf16 GEMM with WMMA, templated on block tile shape.
//   C[M,N] = A[M,K] * B[K,N] (+bias[N]) (+resid) (relu?) -> f32 and/or bf16
//   B is always supplied N-major ("BT"): BT[n][k] = B[k][n], row stride ldb.
//   ctrans: store bf16 C transposed: CbT[n][m] with row stride ldct.
//   Two-level batch: z -> (outer = z/innerB, inner = z%innerB).
// 256 threads = 8 waves, WM x WN wave grid, each wave 2x2 16x16 fragments.
// Tiles: <64,128,2,4> for wide-N GEMMs, <128,64,4,2> for N<=64 GEMMs.
// All call sites guarantee lda/ldb % 8 == 0 and >=16B-aligned bases, so the
// in-bounds fast path uses 16-byte global loads and 16-byte LDS stores.
// Staging is three-way: vector load / vector zero-fill (row OOB, no loads) /
// elementwise K-clamped (only K%32!=0, row known valid -> 32-bit clamp).
// ---------------------------------------------------------------------------
template <int BM, int BN, int WM, int WN>
__global__ __launch_bounds__(256) void k_gemm(
    const bf16_t* __restrict__ A, const bf16_t* __restrict__ BT,
    const float* __restrict__ bias, const float* __restrict__ resid,
    float* __restrict__ Cf, bf16_t* __restrict__ Cb,
    int M, int N, int K, int lda, int ldb, int ldc, int ldct, int ctrans,
    int relu,
    long sAo, long sAi, long sBo, long sBi, long sCo, long sCi, int innerB)
{
  static_assert(BM / (WM * 16) == 2 && BN / (WN * 16) == 2, "2x2 frags per wave");
  __shared__ bf16_t sA[BM * LDSP];    // [m][k]
  __shared__ bf16_t sB[BN * LDSP];    // [n][k]

  int z = blockIdx.z;
  int outer = z / innerB, inner = z - outer * innerB;
  A  += (long)outer * sAo + (long)inner * sAi;
  BT += (long)outer * sBo + (long)inner * sBi;
  long coff = (long)outer * sCo + (long)inner * sCi;

  int tid  = threadIdx.x;
  int wave = tid >> 5, lane = tid & 31;
  int wm = wave % WM, wn = wave / WM;
  int lm = lane & 15, lh = lane >> 4;
  int m0 = blockIdx.x * BM, n0 = blockIdx.y * BN;

  // staging assignments (fixed per thread)
  constexpr int AE  = (BM * 32) / 256;   // A elems per thread (8 or 16)
  constexpr int ATR = 32 / AE;           // threads per A row
  constexpr int BE  = (BN * 32) / 256;   // B elems per thread
  constexpr int BTR = 32 / BE;
  int ar = tid / ATR, ac = (tid % ATR) * AE;
  int bn = tid / BTR, bc = (tid % BTR) * BE;
  bool aRowOk = (m0 + ar) < M;
  bool bRowOk = (n0 + bn) < N;
  long aRow = (long)(m0 + ar) * lda;     // only used when aRowOk
  long bRow = (long)(n0 + bn) * ldb;     // only used when bRowOk

  const uint4 zero4 = {0u, 0u, 0u, 0u};
  v8f acc[2][2] = {};

  for (int kk = 0; kk < K; kk += 32) {
    bool kFull = (kk + 32) <= K;
    // ---- stage A tile BMx32 ----
    if (!aRowOk) {
#pragma unroll
      for (int u = 0; u < AE / 8; ++u)
        *(uint4*)&sA[ar * LDSP + ac + u * 8] = zero4;
    } else if (kFull) {
#pragma unroll
      for (int u = 0; u < AE / 8; ++u)
        *(uint4*)&sA[ar * LDSP + ac + u * 8] =
            *(const uint4*)(A + aRow + kk + ac + u * 8);
      if (kk + 32 < K) __builtin_prefetch(A + aRow + kk + 32 + ac, 0, 1);
    } else {
      const bf16_t* Arow = A + aRow;
#pragma unroll
      for (int j = 0; j < AE; ++j) {
        int gk  = kk + ac + j;
        int gkc = gk < K ? gk : 0;              // 32-bit clamp, branch-free
        bf16_t v = Arow[gkc];
        sA[ar * LDSP + ac + j] = gk < K ? v : (bf16_t)0.0f;
      }
    }
    // ---- stage B tile BNx32 (N-major, contiguous K) ----
    if (!bRowOk) {
#pragma unroll
      for (int u = 0; u < BE / 8; ++u)
        *(uint4*)&sB[bn * LDSP + bc + u * 8] = zero4;
    } else if (kFull) {
#pragma unroll
      for (int u = 0; u < BE / 8; ++u)
        *(uint4*)&sB[bn * LDSP + bc + u * 8] =
            *(const uint4*)(BT + bRow + kk + bc + u * 8);
      if (kk + 32 < K) __builtin_prefetch(BT + bRow + kk + 32 + bc, 0, 1);
    } else {
      const bf16_t* Brow = BT + bRow;
#pragma unroll
      for (int j = 0; j < BE; ++j) {
        int gk  = kk + bc + j;
        int gkc = gk < K ? gk : 0;
        bf16_t v = Brow[gkc];
        sB[bn * LDSP + bc + j] = gk < K ? v : (bf16_t)0.0f;
      }
    }
    __syncthreads();

    // ---- gather fragments per ISA 16-bit A-matrix layout ----
    // lanes 0-15: rows; VGPR v<4 packs K = lh*8 + 2v, v>=4 packs K = 16 + lh*8 + 2(v-4)
    union FragU { unsigned u[8]; v16bf v; };
    FragU af[2], bfr[2];
#pragma unroll
    for (int f = 0; f < 2; ++f) {
      const unsigned short* base =
          (const unsigned short*)sA + (wm * 32 + f * 16 + lm) * LDSP;
#pragma unroll
      for (int v = 0; v < 8; ++v) {
        int k = (v < 4) ? (lh * 8 + 2 * v) : (16 + lh * 8 + 2 * (v - 4));
        af[f].u[v] = *(const unsigned*)(base + k);
      }
    }
#pragma unroll
    for (int g = 0; g < 2; ++g) {
      const unsigned short* base =
          (const unsigned short*)sB + (wn * 32 + g * 16 + lm) * LDSP;
#pragma unroll
      for (int v = 0; v < 8; ++v) {
        int k = (v < 4) ? (lh * 8 + 2 * v) : (16 + lh * 8 + 2 * (v - 4));
        bfr[g].u[v] = *(const unsigned*)(base + k);
      }
    }
#pragma unroll
    for (int f = 0; f < 2; ++f)
#pragma unroll
      for (int g = 0; g < 2; ++g)
        acc[f][g] = __builtin_amdgcn_wmma_f32_16x16x32_bf16(
            false, af[f].v, false, bfr[g].v, (short)0, acc[f][g], false, false);
    __syncthreads();
  }

  // ---- epilogue: bias, residual, relu, f32 / bf16 / transposed-bf16 stores ----
#pragma unroll
  for (int f = 0; f < 2; ++f) {
#pragma unroll
    for (int g = 0; g < 2; ++g) {
#pragma unroll
      for (int r = 0; r < 8; ++r) {
        int gm = m0 + wm * 32 + f * 16 + lh * 8 + r;  // VGPR r: lanes0-15 M=r, lanes16-31 M=r+8
        int gn = n0 + wn * 32 + g * 16 + lm;
        if (gm < M && gn < N) {
          float x = acc[f][g][r];
          if (bias)  x += bias[gn];
          long ci = coff + (long)gm * ldc + gn;
          if (resid) x += resid[ci];
          if (relu)  x = fmaxf(x, 0.0f);
          if (ctrans) {
            if (Cb) Cb[coff + (long)gn * ldct + gm] = (bf16_t)x;
          } else {
            if (Cf) Cf[ci] = x;
            if (Cb) Cb[ci] = (bf16_t)x;
          }
        }
      }
    }
  }
}

// ---------------------------------------------------------------------------
// LayerNorm over last dim D (block per row), optional f32 and bf16 outputs
// ---------------------------------------------------------------------------
__global__ __launch_bounds__(256) void k_layernorm(
    const float* __restrict__ x, const float* __restrict__ g,
    const float* __restrict__ b, float* __restrict__ of,
    bf16_t* __restrict__ ob, int D)
{
  long row = blockIdx.x;
  x += row * (long)D;
  float s = 0.f, s2 = 0.f;
  for (int i = threadIdx.x; i < D; i += 256) {
    float v = x[i]; s += v; s2 += v * v;
  }
#pragma unroll
  for (int o = 16; o; o >>= 1) {            // wave32 reduction
    s  += __shfl_xor(s,  o, 32);
    s2 += __shfl_xor(s2, o, 32);
  }
  __shared__ float sh1[8], sh2[8];
  if ((threadIdx.x & 31) == 0) { sh1[threadIdx.x >> 5] = s; sh2[threadIdx.x >> 5] = s2; }
  __syncthreads();
  s = 0.f; s2 = 0.f;
#pragma unroll
  for (int j = 0; j < 8; ++j) { s += sh1[j]; s2 += sh2[j]; }
  float mean = s / D;
  float var  = s2 / D - mean * mean;
  float inv  = rsqrtf(var + 1e-5f);
  for (int i = threadIdx.x; i < D; i += 256) {
    float v = (x[i] - mean) * inv * g[i] + b[i];
    if (of) of[row * (long)D + i] = v;
    if (ob) ob[row * (long)D + i] = (bf16_t)v;
  }
}

// ---------------------------------------------------------------------------
// Masked softmax over last dim L (block per row). valid keys: j <= maskLen
// (maskLen < 0 => no mask). Applies score scale; emits bf16 probabilities.
// ---------------------------------------------------------------------------
__global__ __launch_bounds__(256) void k_softmax(
    const float* __restrict__ in, bf16_t* __restrict__ out,
    int L, float scale, int maskLen)
{
  long row = blockIdx.x;
  in  += row * (long)L;
  out += row * (long)L;
  __shared__ float sh[8];

  float mx = -3.0e38f;
  for (int i = threadIdx.x; i < L; i += 256) {
    float v = in[i] * scale;
    if (maskLen >= 0 && i > maskLen) v = -3.0e38f;
    mx = fmaxf(mx, v);
  }
#pragma unroll
  for (int o = 16; o; o >>= 1) mx = fmaxf(mx, __shfl_xor(mx, o, 32));
  if ((threadIdx.x & 31) == 0) sh[threadIdx.x >> 5] = mx;
  __syncthreads();
  mx = sh[0];
#pragma unroll
  for (int j = 1; j < 8; ++j) mx = fmaxf(mx, sh[j]);
  __syncthreads();

  float sum = 0.f;
  for (int i = threadIdx.x; i < L; i += 256) {
    float v = in[i] * scale;
    if (maskLen >= 0 && i > maskLen) v = -3.0e38f;
    sum += __expf(v - mx);
  }
#pragma unroll
  for (int o = 16; o; o >>= 1) sum += __shfl_xor(sum, o, 32);
  if ((threadIdx.x & 31) == 0) sh[threadIdx.x >> 5] = sum;
  __syncthreads();
  sum = 0.f;
#pragma unroll
  for (int j = 0; j < 8; ++j) sum += sh[j];
  float rs = 1.f / sum;

  for (int i = threadIdx.x; i < L; i += 256) {
    float v = in[i] * scale;
    if (maskLen >= 0 && i > maskLen) v = -3.0e38f;
    out[i] = (bf16_t)(__expf(v - mx) * rs);
  }
}

// ---------------------------------------------------------------------------
// Encoder input: x[b,s,:] = src[b,:,s] + scale*pe[s,:]  (src is [B,D,S])
// ---------------------------------------------------------------------------
__global__ void k_src_pe(const float* __restrict__ src,
                         const float* __restrict__ pe_scale,
                         float* __restrict__ of, bf16_t* __restrict__ ob)
{
  int bs = blockIdx.x;               // b*SEQ + s
  int b = bs / SEQ, s = bs - b * SEQ;
  float sc = pe_scale[0];
  for (int d = threadIdx.x; d < DMODEL; d += blockDim.x) {
    float v = src[((long)b * DMODEL + d) * SEQ + s] + sc * pe_val(s, d, DMODEL);
    of[(long)bs * DMODEL + d] = v;
    ob[(long)bs * DMODEL + d] = (bf16_t)v;
  }
}

// Decoder input: e[b,t,:] = emb[trg[b,t]] + scale*pe[t,:]
__global__ void k_embed(const int* __restrict__ trg, const float* __restrict__ emb,
                        const float* __restrict__ pe_scale,
                        float* __restrict__ of, bf16_t* __restrict__ ob)
{
  int bt = blockIdx.x;               // b*TDEC + t
  int t = bt % TDEC;
  int tok = trg[bt];
  float sc = pe_scale[0];
  for (int d = threadIdx.x; d < DMODEL; d += blockDim.x) {
    float v = emb[(long)tok * DMODEL + d] + sc * pe_val(t, d, DMODEL);
    of[(long)bt * DMODEL + d] = v;
    ob[(long)bt * DMODEL + d] = (bf16_t)v;
  }
}

__global__ void k_init_trg(int* __restrict__ trg, int n) {
  int i = blockIdx.x * blockDim.x + threadIdx.x;
  if (i < n) trg[i] = 1;
}

// argmax over VOCAB logits for each b; write token into trg[:, pos]
__global__ void k_argmax(const float* __restrict__ logits, int* __restrict__ trg, int pos) {
  int b = threadIdx.x;               // launched with NB threads
  const float* l = logits + (long)b * VOCAB;
  int best = 0; float bv = l[0];
  for (int j = 1; j < VOCAB; ++j) {
    float v = l[j];
    if (v > bv) { bv = v; best = j; } // first max wins, like jnp.argmax
  }
  trg[b * TDEC + pos] = best;
}

__global__ void k_write_out(const int* __restrict__ trg, float* __restrict__ out, int n) {
  int i = blockIdx.x * blockDim.x + threadIdx.x;
  if (i < n) out[i] = (float)trg[i];
}

// fp32 [K,N] weight -> transposed bf16 [N,K] (one-time; writes coalesced)
__global__ void k_cvt_t(const float* __restrict__ s, bf16_t* __restrict__ d, int K, int N) {
  long i = (long)blockIdx.x * blockDim.x + threadIdx.x;   // over N*K (output-linear)
  if (i < (long)K * N) {
    int n = (int)(i / K), k = (int)(i - (long)n * K);
    d[i] = (bf16_t)s[(long)k * N + n];
  }
}

// ---------------------------------------------------------------------------
// Host-side GEMM launcher: pick tile shape by N
// ---------------------------------------------------------------------------
static void gemm(hipStream_t st,
                 const bf16_t* A, int lda, long sAo, long sAi,
                 const bf16_t* BT, int ldb, long sBo, long sBi,
                 const float* bias, const float* resid,
                 float* Cf, bf16_t* Cb, int ldc, long sCo, long sCi,
                 int ldct, int ctrans,
                 int M, int N, int K, int batch, int innerB, int relu)
{
  if (N <= 64) {
    dim3 grid((unsigned)((M + 127) / 128), (unsigned)((N + 63) / 64), (unsigned)batch);
    k_gemm<128, 64, 4, 2><<<grid, 256, 0, st>>>(
        A, BT, bias, resid, Cf, Cb, M, N, K, lda, ldb, ldc, ldct, ctrans, relu,
        sAo, sAi, sBo, sBi, sCo, sCi, innerB);
  } else {
    dim3 grid((unsigned)((M + 63) / 64), (unsigned)((N + 127) / 128), (unsigned)batch);
    k_gemm<64, 128, 2, 4><<<grid, 256, 0, st>>>(
        A, BT, bias, resid, Cf, Cb, M, N, K, lda, ldb, ldc, ldct, ctrans, relu,
        sAo, sAi, sBo, sBi, sCo, sCi, innerB);
  }
}

// ---------------------------------------------------------------------------
// kernel_launch
// ---------------------------------------------------------------------------
extern "C" void kernel_launch(void* const* d_in, const int* in_sizes, int n_in,
                              void* d_out, int out_size, void* d_ws, size_t ws_size,
                              hipStream_t stream)
{
  (void)in_sizes; (void)n_in; (void)out_size; (void)ws_size;

  // ---- input unpacking (setup_inputs dict order, recursively flattened) ----
  const float* src      = (const float*)d_in[0];
  const float* pe_scale = (const float*)d_in[1];
  const float* emb      = (const float*)d_in[2];
  // enc layer: Wq Wk Wv Wo bq bk bv bo W1 b1 W2 b2 ln1g ln1b ln2g ln2b  (16)
  const float* encP[LENC][16];
  for (int l = 0; l < LENC; ++l)
    for (int j = 0; j < 16; ++j)
      encP[l][j] = (const float*)d_in[3 + l * 16 + j];
  const float* enc_ln_g = (const float*)d_in[51];
  const float* enc_ln_b = (const float*)d_in[52];
  // dec layer: saWq..saWo sab* caWq..caWo cab* W1 b1 W2 b2 ln1g ln1b ln2g ln2b ln3g ln3b (26)
  const float* decP[LDEC][26];
  for (int l = 0; l < LDEC; ++l)
    for (int j = 0; j < 26; ++j)
      decP[l][j] = (const float*)d_in[53 + l * 26 + j];
  const float* dec_ln_g = (const float*)d_in[131];
  const float* dec_ln_b = (const float*)d_in[132];
  const float* fcW      = (const float*)d_in[133];
  // d_in[134] = seq_size (48, fixed by setup; hardcoded as TDEC)

  const long MS = (long)NB * SEQ;    // 32768 encoder tokens
  const long MT = (long)NB * TDEC;   // 3072 decoder tokens

  // ---- deterministic workspace carving ----
  size_t off = 0;
  auto carve = [&](size_t bytes) -> void* {
    void* p = (char*)d_ws + off;
    off += (bytes + 255) & ~(size_t)255;
    return p;
  };

  float*  x_f   = (float*) carve(MS * DMODEL * 4);
  bf16_t* x_b   = (bf16_t*)carve(MS * DMODEL * 2);
  float*  y_f   = (float*) carve(MS * DMODEL * 4);
  bf16_t* q_b   = (bf16_t*)carve(MS * DMODEL * 2);
  bf16_t* k_b   = (bf16_t*)carve(MS * DMODEL * 2);
  bf16_t* vT_b  = (bf16_t*)carve(MS * DMODEL * 2);   // [D][B*S] transposed V
  bf16_t* ao_b  = (bf16_t*)carve(MS * DMODEL * 2);
  bf16_t* ffh_b = (bf16_t*)carve(MS * DFFN * 2);
  float*  sc_f  = (float*) carve((size_t)64 * SEQ * SEQ * 4);   // 64-bh score chunk
  bf16_t* pr_b  = (bf16_t*)carve((size_t)64 * SEQ * SEQ * 2);
  bf16_t* mem_b = (bf16_t*)carve(MS * DMODEL * 2);
  bf16_t* kmem[LDEC], *vmemT[LDEC];
  for (int l = 0; l < LDEC; ++l) {
    kmem[l]  = (bf16_t*)carve(MS * DMODEL * 2);
    vmemT[l] = (bf16_t*)carve(MS * DMODEL * 2);      // [D][B*S]
  }
  float*  dx_f   = (float*) carve(MT * DMODEL * 4);
  bf16_t* dx_b   = (bf16_t*)carve(MT * DMODEL * 2);
  float*  dy_f   = (float*) carve(MT * DMODEL * 4);
  bf16_t* dq_b   = (bf16_t*)carve(MT * DMODEL * 2);
  bf16_t* dk_b   = (bf16_t*)carve(MT * DMODEL * 2);
  bf16_t* dvT_b  = (bf16_t*)carve(MT * DMODEL * 2);  // [D][B*T]
  bf16_t* dao_b  = (bf16_t*)carve(MT * DMODEL * 2);
  bf16_t* dffh_b = (bf16_t*)carve(MT * DFFN * 2);
  float*  dsc_f  = (float*) carve((size_t)NB * NHEAD * TDEC * SEQ * 4); // fits self & cross
  bf16_t* dpr_b  = (bf16_t*)carve((size_t)NB * NHEAD * TDEC * SEQ * 2);
  bf16_t* dh_b   = (bf16_t*)carve(MT * DMODEL * 2);
  float*  logits = (float*) carve((size_t)NB * VOCAB * 4);
  int*    trg    = (int*)   carve((size_t)NB * TDEC * 4);

  // ---- one-time weight fp32 -> transposed bf16 conversion (B is N-major) ----
  auto cvt_t = [&](const float* s, int K, int N) -> bf16_t* {
    long n = (long)K * N;
    bf16_t* d = (bf16_t*)carve((size_t)n * 2);
    k_cvt_t<<<(unsigned)((n + 255) / 256), 256, 0, stream>>>(s, d, K, N);
    return d;
  };
  bf16_t* encW[LENC][6];  // WqT WkT WvT WoT W1T W2T
  for (int l = 0; l < LENC; ++l) {
    for (int j = 0; j < 4; ++j) encW[l][j] = cvt_t(encP[l][j], DMODEL, DMODEL);
    encW[l][4] = cvt_t(encP[l][8],  DMODEL, DFFN);   // [DFFN][DMODEL]
    encW[l][5] = cvt_t(encP[l][10], DFFN, DMODEL);   // [DMODEL][DFFN]
  }
  bf16_t* decW[LDEC][10]; // saW*T caW*T W1T W2T
  for (int l = 0; l < LDEC; ++l) {
    for (int j = 0; j < 4; ++j) decW[l][j]     = cvt_t(decP[l][j],     DMODEL, DMODEL);
    for (int j = 0; j < 4; ++j) decW[l][4 + j] = cvt_t(decP[l][8 + j], DMODEL, DMODEL);
    decW[l][8] = cvt_t(decP[l][16], DMODEL, DFFN);
    decW[l][9] = cvt_t(decP[l][18], DFFN, DMODEL);
  }
  bf16_t* fcT = cvt_t(fcW, DMODEL, VOCAB);           // [VOCAB][DMODEL]

  // ---- batched multi-head attention: two GEMMs + masked softmax ----
  // Q:[nb,Sq,D] K:[nb,Sk,D] (head slice at h*HDIM), VT:[D][nb*Sk], O like Q
  auto attention = [&](const bf16_t* Q, const bf16_t* Kb, const bf16_t* VT, long ldvt,
                       bf16_t* Ob, int Sq, int Sk, int nb, float* scf, bf16_t* prb,
                       int maskLen, int chunkB) {
    for (int c0 = 0; c0 < nb; c0 += chunkB) {
      const bf16_t* Qc = Q  + (long)c0 * Sq * DMODEL;
      const bf16_t* Kc = Kb + (long)c0 * Sk * DMODEL;
      const bf16_t* Vc = VT + (long)c0 * Sk;          // column offset per batch
      bf16_t*       Oc = Ob + (long)c0 * Sq * DMODEL;
      // scores[bh,Sq,Sk] = Qh * Kh^T   (K buffer is already N-major here)
      gemm(stream,
           Qc, DMODEL, (long)Sq * DMODEL, HDIM,
           Kc, DMODEL, (long)Sk * DMODEL, HDIM,
           nullptr, nullptr, scf, nullptr, Sk,
           (long)NHEAD * Sq * Sk, (long)Sq * Sk, 0, 0,
           Sq, Sk, HDIM, chunkB * NHEAD, NHEAD, 0);
      k_softmax<<<(unsigned)((long)chunkB * NHEAD * Sq), 256, 0, stream>>>(
          scf, prb, Sk, 0.125f /*1/sqrt(64)*/, maskLen);
      // O[bh,Sq,hd] = P * Vh   (V^T gives N-major B with contiguous K)
      gemm(stream,
           prb, Sk, (long)NHEAD * Sq * Sk, (long)Sq * Sk,
           Vc, (int)ldvt, (long)Sk, (long)HDIM * ldvt,
           nullptr, nullptr, nullptr, Oc, DMODEL,
           (long)Sq * DMODEL, HDIM, 0, 0,
           Sq, HDIM, Sk, chunkB * NHEAD, NHEAD, 0);
    }
  };

  // =========================== ENCODER ===========================
  k_src_pe<<<(unsigned)MS, 256, 0, stream>>>(src, pe_scale, x_f, x_b);

  for (int l = 0; l < LENC; ++l) {
    const float* const* p = encP[l];
    bf16_t* const* w = encW[l];
    gemm(stream, x_b, DMODEL,0,0, w[0], DMODEL,0,0, p[4], nullptr, nullptr, q_b, DMODEL,0,0, 0,0, (int)MS, DMODEL, DMODEL, 1,1,0);
    gemm(stream, x_b, DMODEL,0,0, w[1], DMODEL,0,0, p[5], nullptr, nullptr, k_b, DMODEL,0,0, 0,0, (int)MS, DMODEL, DMODEL, 1,1,0);
    gemm(stream, x_b, DMODEL,0,0, w[2], DMODEL,0,0, p[6], nullptr, nullptr, vT_b, DMODEL,0,0, (int)MS,1, (int)MS, DMODEL, DMODEL, 1,1,0);
    attention(q_b, k_b, vT_b, MS, ao_b, SEQ, SEQ, NB, sc_f, pr_b, -1, 8);
    gemm(stream, ao_b, DMODEL,0,0, w[3], DMODEL,0,0, p[7], x_f, y_f, nullptr, DMODEL,0,0, 0,0, (int)MS, DMODEL, DMODEL, 1,1,0);
    k_layernorm<<<(unsigned)MS, 256, 0, stream>>>(y_f, p[12], p[13], x_f, x_b, DMODEL);
    gemm(stream, x_b, DMODEL,0,0, w[4], DMODEL,0,0, p[9], nullptr, nullptr, ffh_b, DFFN,0,0, 0,0, (int)MS, DFFN, DMODEL, 1,1,/*relu=*/1);
    gemm(stream, ffh_b, DFFN,0,0, w[5], DFFN,0,0, p[11], x_f, y_f, nullptr, DMODEL,0,0, 0,0, (int)MS, DMODEL, DFFN, 1,1,0);
    k_layernorm<<<(unsigned)MS, 256, 0, stream>>>(y_f, p[14], p[15], x_f, x_b, DMODEL);
  }
  // encoder final LayerNorm -> mem (bf16 only; consumed as GEMM operand)
  k_layernorm<<<(unsigned)MS, 256, 0, stream>>>(x_f, enc_ln_g, enc_ln_b, nullptr, mem_b, DMODEL);

  // hoist step-invariant cross-attention K/V projections of mem (per layer)
  for (int l = 0; l < LDEC; ++l) {
    gemm(stream, mem_b, DMODEL,0,0, decW[l][5], DMODEL,0,0, decP[l][13], nullptr, nullptr, kmem[l], DMODEL,0,0, 0,0, (int)MS, DMODEL, DMODEL, 1,1,0);
    gemm(stream, mem_b, DMODEL,0,0, decW[l][6], DMODEL,0,0, decP[l][14], nullptr, nullptr, vmemT[l], DMODEL,0,0, (int)MS,1, (int)MS, DMODEL, DMODEL, 1,1,0);
  }

  // =========================== DECODER ===========================
  k_init_trg<<<(NB * TDEC + 255) / 256, 256, 0, stream>>>(trg, NB * TDEC);

  for (int i = 0; i < TDEC - 1; ++i) {
    k_embed<<<(unsigned)MT, 256, 0, stream>>>(trg, emb, pe_scale, dx_f, dx_b);

    for (int l = 0; l < LDEC; ++l) {
      const float* const* p = decP[l];
      bf16_t* const* w = decW[l];
      // --- masked self-attention (keys valid for j <= i) ---
      gemm(stream, dx_b, DMODEL,0,0, w[0], DMODEL,0,0, p[4], nullptr, nullptr, dq_b, DMODEL,0,0, 0,0, (int)MT, DMODEL, DMODEL, 1,1,0);
      gemm(stream, dx_b, DMODEL,0,0, w[1], DMODEL,0,0, p[5], nullptr, nullptr, dk_b, DMODEL,0,0, 0,0, (int)MT, DMODEL, DMODEL, 1,1,0);
      gemm(stream, dx_b, DMODEL,0,0, w[2], DMODEL,0,0, p[6], nullptr, nullptr, dvT_b, DMODEL,0,0, (int)MT,1, (int)MT, DMODEL, DMODEL, 1,1,0);
      attention(dq_b, dk_b, dvT_b, MT, dao_b, TDEC, TDEC, NB, dsc_f, dpr_b, i, NB);
      gemm(stream, dao_b, DMODEL,0,0, w[3], DMODEL,0,0, p[7], dx_f, dy_f, nullptr, DMODEL,0,0, 0,0, (int)MT, DMODEL, DMODEL, 1,1,0);
      k_layernorm<<<(unsigned)MT, 256, 0, stream>>>(dy_f, p[20], p[21], dx_f, dx_b, DMODEL);
      // --- cross-attention (K/V hoisted) ---
      gemm(stream, dx_b, DMODEL,0,0, w[4], DMODEL,0,0, p[12], nullptr, nullptr, dq_b, DMODEL,0,0, 0,0, (int)MT, DMODEL, DMODEL, 1,1,0);
      attention(dq_b, kmem[l], vmemT[l], MS, dao_b, TDEC, SEQ, NB, dsc_f, dpr_b, -1, NB);
      gemm(stream, dao_b, DMODEL,0,0, w[7], DMODEL,0,0, p[15], dx_f, dy_f, nullptr, DMODEL,0,0, 0,0, (int)MT, DMODEL, DMODEL, 1,1,0);
      k_layernorm<<<(unsigned)MT, 256, 0, stream>>>(dy_f, p[22], p[23], dx_f, dx_b, DMODEL);
      // --- FFN ---
      gemm(stream, dx_b, DMODEL,0,0, w[8], DMODEL,0,0, p[17], nullptr, nullptr, dffh_b, DFFN,0,0, 0,0, (int)MT, DFFN, DMODEL, 1,1,/*relu=*/1);
      gemm(stream, dffh_b, DFFN,0,0, w[9], DFFN,0,0, p[19], dx_f, dy_f, nullptr, DMODEL,0,0, 0,0, (int)MT, DMODEL, DFFN, 1,1,0);
      k_layernorm<<<(unsigned)MT, 256, 0, stream>>>(dy_f, p[24], p[25], dx_f, dx_b, DMODEL);
    }
    // decoder final LN -> dh (bf16), logits for row i only, argmax -> trg[:, i+1]
    k_layernorm<<<(unsigned)MT, 256, 0, stream>>>(dx_f, dec_ln_g, dec_ln_b, nullptr, dh_b, DMODEL);
    gemm(stream, dh_b + (long)i * DMODEL, TDEC * DMODEL, 0,0,
         fcT, DMODEL, 0,0, nullptr, nullptr, logits, nullptr, VOCAB, 0,0, 0,0,
         NB, VOCAB, DMODEL, 1,1,0);
    k_argmax<<<1, NB, 0, stream>>>(logits, trg, i + 1);
  }

  k_write_out<<<(NB * TDEC + 255) / 256, 256, 0, stream>>>(trg, (float*)d_out, NB * TDEC);
}
